// QLoRABigNet_42528766165136
// MI455X (gfx1250) — compile-verified, matrix operand order checked by reading, not run on hardware
//
#include <hip/hip_runtime.h>
#include <hip/hip_bf16.h>

#define DIM     1024
#define BATCH   8192
#define NLAYERS 18
#define NBLOCKS 6
#define LORA    20
#define GROUP   16
#define EPS     1e-5f

typedef __attribute__((ext_vector_type(16))) __bf16 v16bf;
typedef __attribute__((ext_vector_type(8)))  float  v8f;

union FragU {
    uint4 u[2];
    v16bf v;
};

// ---------------------------------------------------------------------------
// Async DMA: copy 32 bytes (16 bf16) global -> LDS without touching VGPR data.
// Per ISA 08_async_tensor: INST_OFFSET is added to BOTH the LDS and global
// addresses, so one base pair serves both 16B chunks. Tracked by ASYNCcnt.
// LDS address = low 32 bits of the generic pointer (LDS aperture truncation).
// ---------------------------------------------------------------------------
__device__ __forceinline__ void qbn_async_cp32(const __bf16* g, const __bf16* l)
{
    unsigned lds = (unsigned)(uintptr_t)l;
    asm volatile("global_load_async_to_lds_b128 %0, %1, off"
                 :: "v"(lds), "v"(g) : "memory");
    asm volatile("global_load_async_to_lds_b128 %0, %1, off offset:16"
                 :: "v"(lds), "v"(g) : "memory");
}

__device__ __forceinline__ void qbn_wait_async0()
{
    asm volatile("s_wait_asynccnt 0" ::: "memory");
}

// ---------------------------------------------------------------------------
// Prep: W_eff[l][o][i] = qw*scale + sum_r B[o][r]*A[r][i], stored bf16.
// ---------------------------------------------------------------------------
__global__ __launch_bounds__(256) void qbn_prep_weights(
    const int* __restrict__ qw, const float* __restrict__ scales,
    const float* __restrict__ lora_a, const float* __restrict__ lora_b,
    __bf16* __restrict__ weff)
{
    size_t idx = (size_t)blockIdx.x * 256 + threadIdx.x;   // < 18*1024*1024
    int l   = (int)(idx >> 20);
    int rem = (int)(idx & ((1u << 20) - 1));
    int o   = rem >> 10;
    int i   = rem & (DIM - 1);

    float w = (float)qw[idx] *
              scales[((size_t)l * DIM + o) * (DIM / GROUP) + (i / GROUP)];

    const float* A = lora_a + (size_t)l * LORA * DIM;        // [LORA][DIM]
    const float* B = lora_b + ((size_t)l * DIM + o) * LORA;  // [DIM][LORA]
#pragma unroll
    for (int r = 0; r < LORA; ++r)
        w = fmaf(B[r], A[(size_t)r * DIM + i], w);

    weff[idx] = (__bf16)w;
}

// ---------------------------------------------------------------------------
// Init: h = x (f32), act = bf16(x)
// ---------------------------------------------------------------------------
__global__ __launch_bounds__(256) void qbn_init(
    const float* __restrict__ x, float* __restrict__ h, __bf16* __restrict__ act)
{
    size_t idx = (size_t)blockIdx.x * 256 + threadIdx.x;
    float v = x[idx];
    h[idx] = v;
    act[idx] = (__bf16)v;
}

// ---------------------------------------------------------------------------
// GEMM: out[N, DIM] = act[N, DIM] @ W^T + bias (+ReLU), W row-major [out][in].
// Workgroup = 256 threads = 8 wave32s; tile 128x128; K-step 32.
// Wave grid 4(M) x 2(N): each wave computes 32x64 via 2x4 WMMA 16x16 tiles.
// Double-buffered LDS filled by async global->LDS DMA (ASYNCcnt pipelined).
// ---------------------------------------------------------------------------
#define BM 128
#define BN 128
#define BK 32

__global__ __launch_bounds__(256) void qbn_gemm_wmma(
    const __bf16* __restrict__ act, const __bf16* __restrict__ W,
    const float* __restrict__ bias, float* __restrict__ out32,
    __bf16* __restrict__ actout, int relu)
{
    __shared__ __align__(16) __bf16 As[2][BM * BK];
    __shared__ __align__(16) __bf16 Bs[2][BN * BK];

    const int tid  = threadIdx.x;
    const int wave = tid >> 5;
    const int lane = tid & 31;
    const int wm   = wave & 3;        // 0..3  -> M offset 32*wm
    const int wn   = wave >> 2;       // 0..1  -> N offset 64*wn
    const int half = lane >> 4;       // lane group
    const int lr   = lane & 15;

    const int rowA0 = blockIdx.x * BM;   // batch rows
    const int colB0 = blockIdx.y * BN;   // output features

    // tile loading: each thread DMAs 16 bf16 (32B) of A and of B per K-step
    const int lrow = tid >> 1;           // 0..127
    const int lcol = (tid & 1) * 16;     // 0 or 16
    const __bf16* gA = act + (size_t)(rowA0 + lrow) * DIM + lcol;
    const __bf16* gB = W   + (size_t)(colB0 + lrow) * DIM + lcol;
    const int ldst = lrow * BK + lcol;

    v8f acc[2][4];
#pragma unroll
    for (int mi = 0; mi < 2; ++mi)
#pragma unroll
        for (int ni = 0; ni < 4; ++ni)
            acc[mi][ni] = (v8f)0.0f;

    // prologue: DMA tile 0 into buffer 0
    qbn_async_cp32(gA, &As[0][ldst]);
    qbn_async_cp32(gB, &Bs[0][ldst]);
    qbn_wait_async0();
    __syncthreads();

    int buf = 0;
#pragma unroll 1
    for (int k0 = 0; k0 < DIM; k0 += BK) {
        const bool more = (k0 + BK) < DIM;
        if (more) {       // overlap next tile's DMA with this tile's WMMAs
            qbn_async_cp32(gA + (k0 + BK), &As[buf ^ 1][ldst]);
            qbn_async_cp32(gB + (k0 + BK), &Bs[buf ^ 1][ldst]);
        }

        // Fragment K split per ISA 16-bit layout: lanes 0-15 hold K 0-7/16-23,
        // lanes 16-31 hold K 8-15/24-31.
        const int kb = half * 8;
        const __bf16* Ab = As[buf];
        const __bf16* Bb = Bs[buf];

        v16bf afrag[2];
#pragma unroll
        for (int mi = 0; mi < 2; ++mi) {
            int r = wm * 32 + mi * 16 + lr;
            FragU f;
            f.u[0] = *(const uint4*)(Ab + r * BK + kb);
            f.u[1] = *(const uint4*)(Ab + r * BK + kb + 16);
            afrag[mi] = f.v;
        }
        v16bf bfrag[4];
#pragma unroll
        for (int ni = 0; ni < 4; ++ni) {
            int c = wn * 64 + ni * 16 + lr;
            FragU f;
            f.u[0] = *(const uint4*)(Bb + c * BK + kb);
            f.u[1] = *(const uint4*)(Bb + c * BK + kb + 16);
            bfrag[ni] = f.v;
        }

#pragma unroll
        for (int mi = 0; mi < 2; ++mi)
#pragma unroll
            for (int ni = 0; ni < 4; ++ni)
                acc[mi][ni] = __builtin_amdgcn_wmma_f32_16x16x32_bf16(
                    false, afrag[mi], false, bfrag[ni],
                    (short)0, acc[mi][ni], false, false);

        if (more) qbn_wait_async0();   // next tile landed in LDS
        __syncthreads();               // everyone done reading current buffer
        buf ^= 1;
    }

    // Epilogue: C layout (16x16 f32): VGPR v, lanes 0-15 -> M=v, lanes 16-31 -> M=8+v
#pragma unroll
    for (int mi = 0; mi < 2; ++mi) {
#pragma unroll
        for (int ni = 0; ni < 4; ++ni) {
            const int col = colB0 + wn * 64 + ni * 16 + lr;
            const float bv = bias[col];
            const int rbase = rowA0 + wm * 32 + mi * 16 + half * 8;
#pragma unroll
            for (int v = 0; v < 8; ++v) {
                float y = acc[mi][ni][v] + bv;
                if (relu) y = fmaxf(y, 0.0f);
                size_t o = (size_t)(rbase + v) * DIM + col;
                out32[o]  = y;
                actout[o] = (__bf16)y;
            }
        }
    }
}

// ---------------------------------------------------------------------------
// Residual + optional LayerNorm, one workgroup per row (256 thr x 4 cols).
// h = h + r ; if doLN: h = LN(h)*w + b.  Writes h (f32) and bf16 act.
// ---------------------------------------------------------------------------
__global__ __launch_bounds__(256) void qbn_residual_ln(
    float* __restrict__ h, const float* __restrict__ r,
    const float* __restrict__ lnw, const float* __restrict__ lnb,
    __bf16* __restrict__ act, int doLN)
{
    __shared__ float s1[256];
    __shared__ float s2[256];
    const int tid = threadIdx.x;
    const size_t base = (size_t)blockIdx.x * DIM;

    float v4[4];
    float s = 0.f, q = 0.f;
#pragma unroll
    for (int i = 0; i < 4; ++i) {
        int c = tid + i * 256;
        float v = h[base + c] + r[base + c];
        v4[i] = v;
        s += v;
        q += v * v;
    }
    s1[tid] = s;
    s2[tid] = q;
    __syncthreads();
    for (int off = 128; off > 0; off >>= 1) {
        if (tid < off) {
            s1[tid] += s1[tid + off];
            s2[tid] += s2[tid + off];
        }
        __syncthreads();
    }
    const float mean = s1[0] * (1.0f / DIM);
    const float var  = s2[0] * (1.0f / DIM) - mean * mean;
    const float rstd = rsqrtf(var + EPS);

#pragma unroll
    for (int i = 0; i < 4; ++i) {
        int c = tid + i * 256;
        float v = v4[i];
        float o = doLN ? (v - mean) * rstd * lnw[c] + lnb[c] : v;
        h[base + c]   = o;
        act[base + c] = (__bf16)o;
    }
}

// ---------------------------------------------------------------------------
extern "C" void kernel_launch(void* const* d_in, const int* in_sizes, int n_in,
                              void* d_out, int out_size, void* d_ws, size_t ws_size,
                              hipStream_t stream)
{
    const float* x      = (const float*)d_in[0];
    const int*   qw     = (const int*)  d_in[1];
    const float* scales = (const float*)d_in[2];
    const float* bias   = (const float*)d_in[3];
    const float* lora_a = (const float*)d_in[4];
    const float* lora_b = (const float*)d_in[5];
    const float* ln_w   = (const float*)d_in[6];
    const float* ln_b   = (const float*)d_in[7];
    float* h = (float*)d_out;                       // [BATCH][DIM] fp32 state

    // workspace carve-up
    char* wsp = (char*)d_ws;
    const size_t weff_bytes = (size_t)NLAYERS * DIM * DIM * sizeof(__bf16); // 37.75MB
    const size_t act_bytes  = (size_t)BATCH * DIM * sizeof(__bf16);         // 16.78MB
    __bf16* weff = (__bf16*)wsp;
    __bf16* actA = (__bf16*)(wsp + weff_bytes);
    __bf16* actB = (__bf16*)(wsp + weff_bytes + act_bytes);
    float*  r32  = (float*) (wsp + weff_bytes + 2 * act_bytes);             // 33.55MB

    // 1) build effective bf16 weights (dequant + LoRA fold) for all layers
    {
        size_t total = (size_t)NLAYERS * DIM * DIM;
        qbn_prep_weights<<<(unsigned)(total / 256), 256, 0, stream>>>(
            qw, scales, lora_a, lora_b, weff);
    }

    // 2) h = x, actA = bf16(x)
    {
        size_t total = (size_t)BATCH * DIM;
        qbn_init<<<(unsigned)(total / 256), 256, 0, stream>>>(x, h, actA);
    }

    // 3) 6 blocks x 3 QLoRA layers + residual/LN
    dim3 ggrid(BATCH / BM, DIM / BN);   // 64 x 8
    __bf16* ain = actA;
    __bf16* aout = actB;
    for (int b = 0; b < NBLOCKS; ++b) {
        for (int j = 0; j < 3; ++j) {
            int li = b * 3 + j;
            qbn_gemm_wmma<<<ggrid, 256, 0, stream>>>(
                ain, weff + (size_t)li * DIM * DIM, bias + (size_t)li * DIM,
                r32, aout, (j < 2) ? 1 : 0);
            __bf16* t = ain; ain = aout; aout = t;
        }
        int doLN = (b < NBLOCKS - 1) ? 1 : 0;
        const float* lw = ln_w + (size_t)(doLN ? b : 0) * DIM;
        const float* lb = ln_b + (size_t)(doLN ? b : 0) * DIM;
        // write post-residual activation into `ain` for the next block's GEMM
        qbn_residual_ln<<<BATCH, 256, 0, stream>>>(h, r32, lw, lb, ain, doLN);
    }
    (void)in_sizes; (void)n_in; (void)out_size; (void)ws_size;
}
